// MultiheadAttention_31207232373271
// MI455X (gfx1250) — matrix operand order repaired and checked by reading, not compiled, not run on hardware
//
#include <hip/hip_runtime.h>
#include <hip/hip_bf16.h>

// ---------------- types ----------------
typedef __bf16 bf16_t;
typedef __attribute__((ext_vector_type(16))) __bf16 v16bf;
typedef __attribute__((ext_vector_type(8)))  __bf16 bh8;
typedef __attribute__((ext_vector_type(8)))  float  v8f;

#define NHEAD  12
#define DHEAD  64
#define DMODEL 768
#define SEQ    2048
#define BATCH  2
#define SPAN2  1024                  // 2*ATT_SPAN
#define SCALE_F 0.07216878364870323f // 1/sqrt(64*3)

// ---------------- WMMA helpers (CDNA5 layouts, ISA 7.12.2) ----------------
__device__ __forceinline__ v8f wmma_bf16(v16bf a, v16bf b, v8f c) {
  return __builtin_amdgcn_wmma_f32_16x16x32_bf16(false, a, false, b, (short)0, c,
                                                 false, false);
}

// A fragment 16x32 (MxK): row = lane&15; elems 0..7 -> K=(lane>>4)*8+e,
// elems 8..15 -> K=16+(lane>>4)*8+(e-8). base points at tile [row][k=0].
__device__ __forceinline__ v16bf frag_a(const bf16_t* base, int stride, int lane) {
  const int row = lane & 15;
  const int kb  = (lane >> 4) * 8;
  const bf16_t* p = base + (size_t)row * stride;
  bh8 lo = *(const bh8*)(p + kb);
  bh8 hi = *(const bh8*)(p + 16 + kb);
  v16bf r;
#pragma unroll
  for (int e = 0; e < 8; ++e) { r[e] = lo[e]; r[e + 8] = hi[e]; }
  return r;
}

// B fragment 32x16 (KxN) from a tile stored as [n][k]: col = lane&15,
// elems 0..7 -> K=(lane>>4)*8+e, elems 8..15 -> K=16+(lane>>4)*8+(e-8).
__device__ __forceinline__ v16bf frag_b(const bf16_t* base, int stride, int lane) {
  const int col = lane & 15;
  const int kb  = (lane >> 4) * 8;
  const bf16_t* p = base + (size_t)col * stride + kb;
  bh8 lo = *(const bh8*)(p);
  bh8 hi = *(const bh8*)(p + 16);
  v16bf r;
#pragma unroll
  for (int e = 0; e < 8; ++e) { r[e] = lo[e]; r[e + 8] = hi[e]; }
  return r;
}

__device__ __forceinline__ void cvt_store8(bf16_t* dst, float4 a, float4 b) {
  bh8 v;
  v[0] = (bf16_t)a.x; v[1] = (bf16_t)a.y; v[2] = (bf16_t)a.z; v[3] = (bf16_t)a.w;
  v[4] = (bf16_t)b.x; v[5] = (bf16_t)b.y; v[6] = (bf16_t)b.z; v[7] = (bf16_t)b.w;
  *(bh8*)dst = v;
}

// ---------------- CDNA5 async DMA: global -> LDS, 64B per lane ----------------
// LDS generic address low 32 bits are the LDS byte offset (ISA 10.2 aperture map).
// INST_OFFSET is added to BOTH the LDS and global addresses (ISA 08 §4.4), and the
// 64 bytes are contiguous in both spaces (within one tile row), so 4x B128 works.
__device__ __forceinline__ void async_copy64(const void* lds_ptr, const void* gptr) {
  unsigned lds = (unsigned)(unsigned long long)lds_ptr;
  unsigned long long ga = (unsigned long long)gptr;
  asm volatile(
      "global_load_async_to_lds_b128 %0, %1, off\n\t"
      "global_load_async_to_lds_b128 %0, %1, off offset:16\n\t"
      "global_load_async_to_lds_b128 %0, %1, off offset:32\n\t"
      "global_load_async_to_lds_b128 %0, %1, off offset:48"
      :: "v"(lds), "v"(ga)
      : "memory");
}
__device__ __forceinline__ void wait_async0() {
  asm volatile("s_wait_asynccnt 0" ::: "memory");
}

// ---------------- Kernel 1: projection GEMM  C[M,N](bf16) = (A f32 @ W^T + b1 + b2)*alpha
// Block: 128 thr (4 waves). Tile 64x64, K-step 32. Wave w owns N cols 16w..16w+15.
__global__ __launch_bounds__(128)
void gemm_proj_kernel(const float* __restrict__ A, int lda,
                      const float* __restrict__ W, int ldb,
                      const float* __restrict__ bias1,
                      const float* __restrict__ bias2,
                      float alpha, bf16_t* __restrict__ C, int ldc, int K) {
  __shared__ __align__(16) bf16_t As[64][40];
  __shared__ __align__(16) bf16_t Bs[64][40];
  const int m0 = blockIdx.x * 64, n0 = blockIdx.y * 64;
  const int t = threadIdx.x, lane = t & 31, wave = t >> 5;
  const int lrow = t >> 1, lcb = (t & 1) * 16;
  v8f acc[4] = {};
  for (int k0 = 0; k0 < K; k0 += 32) {
    __syncthreads();
    {  // stage tiles with in-flight f32 -> bf16 conversion
      const float4* ga = (const float4*)(A + (size_t)(m0 + lrow) * lda + k0 + lcb);
      cvt_store8(&As[lrow][lcb],     ga[0], ga[1]);
      cvt_store8(&As[lrow][lcb + 8], ga[2], ga[3]);
      const float4* gb = (const float4*)(W + (size_t)(n0 + lrow) * ldb + k0 + lcb);
      cvt_store8(&Bs[lrow][lcb],     gb[0], gb[1]);
      cvt_store8(&Bs[lrow][lcb + 8], gb[2], gb[3]);
    }
    __syncthreads();
    // batch all fragment loads, then issue the WMMA chain
    v16bf bfr = frag_b(&Bs[16 * wave][0], 40, lane);
    v16bf a0 = frag_a(&As[0][0],  40, lane);
    v16bf a1 = frag_a(&As[16][0], 40, lane);
    v16bf a2 = frag_a(&As[32][0], 40, lane);
    v16bf a3 = frag_a(&As[48][0], 40, lane);
    acc[0] = wmma_bf16(a0, bfr, acc[0]);
    acc[1] = wmma_bf16(a1, bfr, acc[1]);
    acc[2] = wmma_bf16(a2, bfr, acc[2]);
    acc[3] = wmma_bf16(a3, bfr, acc[3]);
  }
  const int colg = n0 + 16 * wave + (lane & 15);
  float badd = 0.0f;
  if (bias1) badd += bias1[colg];
  if (bias2) badd += bias2[colg];
#pragma unroll
  for (int mt = 0; mt < 4; ++mt)
#pragma unroll
    for (int r = 0; r < 8; ++r) {
      const int rowg = m0 + 16 * mt + ((lane >> 4) * 8) + r;
      C[(size_t)rowg * ldc + colg] = (bf16_t)((acc[mt][r] + badd) * alpha);
    }
}

// ---------------- Kernel 2: per-head rel-position GEMM
// C[z][S,1024](bf16) = A_head[S,64] @ Pos_head[1024,64]^T   (z = b*12+n), K=64.
// Tiles arrive via async global->LDS DMA (ASYNCcnt).
__global__ __launch_bounds__(128)
void gemm_rel_kernel(const bf16_t* __restrict__ Ah, const bf16_t* __restrict__ Pos,
                     bf16_t* __restrict__ Cout) {
  __shared__ __align__(16) bf16_t As[64][72];
  __shared__ __align__(16) bf16_t Bs[64][72];
  const int z = blockIdx.z, b = z / NHEAD, n = z % NHEAD;
  const bf16_t* Abase = Ah + ((size_t)b * SEQ) * DMODEL + n * DHEAD;
  const bf16_t* Bbase = Pos + n * DHEAD;
  const int m0 = blockIdx.x * 64, n0 = blockIdx.y * 64;
  const int t = threadIdx.x, lane = t & 31, wave = t >> 5;
  const int lrow = t >> 1, lcb = (t & 1) * 32;
  async_copy64(&As[lrow][lcb], Abase + (size_t)(m0 + lrow) * DMODEL + lcb);
  async_copy64(&Bs[lrow][lcb], Bbase + (size_t)(n0 + lrow) * DMODEL + lcb);
  wait_async0();
  __syncthreads();
  v8f acc[4] = {};
#pragma unroll
  for (int ks = 0; ks < 2; ++ks) {
    v16bf bfr = frag_b(&Bs[16 * wave][ks * 32], 72, lane);
    v16bf a0 = frag_a(&As[0][ks * 32],  72, lane);
    v16bf a1 = frag_a(&As[16][ks * 32], 72, lane);
    v16bf a2 = frag_a(&As[32][ks * 32], 72, lane);
    v16bf a3 = frag_a(&As[48][ks * 32], 72, lane);
    acc[0] = wmma_bf16(a0, bfr, acc[0]);
    acc[1] = wmma_bf16(a1, bfr, acc[1]);
    acc[2] = wmma_bf16(a2, bfr, acc[2]);
    acc[3] = wmma_bf16(a3, bfr, acc[3]);
  }
  bf16_t* Cp = Cout + (size_t)z * SEQ * SPAN2;
  const int colg = n0 + 16 * wave + (lane & 15);
#pragma unroll
  for (int mt = 0; mt < 4; ++mt)
#pragma unroll
    for (int r = 0; r < 8; ++r) {
      const int rowg = m0 + 16 * mt + ((lane >> 4) * 8) + r;
      Cp[(size_t)rowg * SPAN2 + colg] = (bf16_t)acc[mt][r];
    }
}

// ---------------- Kernel 3: fused attention (flash-style, online softmax)
// grid (S/64, B*NHEAD), block 128 (4 waves). Wave w owns query rows 16w..16w+15.
__global__ __launch_bounds__(128)
void attn_kernel(const bf16_t* __restrict__ qh, const bf16_t* __restrict__ kh,
                 const bf16_t* __restrict__ vh, const bf16_t* __restrict__ c2p,
                 const bf16_t* __restrict__ p2c, const float* __restrict__ mask,
                 float* __restrict__ attn_out) {
  __shared__ __align__(16) bf16_t Ks[64][72];  // [key][d]
  __shared__ __align__(16) bf16_t Vt[64][72];  // [d][key]
  __shared__ __align__(16) bf16_t Ps[64][72];  // [qrow][key]
  const int z = blockIdx.y, b = z / NHEAD, n = z % NHEAD;
  const int i0 = blockIdx.x * 64;
  const int t = threadIdx.x, lane = t & 31, wave = t >> 5;
  const int rhalf = lane >> 4;

  // Q fragments for this wave's 16 rows (2 K-steps over d=64), held in regs.
  const bf16_t* qbase = qh + ((size_t)(b * SEQ + i0 + 16 * wave)) * DMODEL + n * DHEAD;
  v16bf qa[2];
  qa[0] = frag_a(qbase, DMODEL, lane);
  qa[1] = frag_a(qbase + 32, DMODEL, lane);

  v8f acc_o[4] = {};
  float mrun[8], lrun[8];
#pragma unroll
  for (int r = 0; r < 8; ++r) { mrun[r] = -1e30f; lrun[r] = 0.0f; }
  const size_t relbase = (size_t)z * SEQ * SPAN2;
  const int lrow = t >> 1, lcb = (t & 1) * 32;

  for (int j0 = 0; j0 < SEQ; j0 += 64) {
    __syncthreads();
    {  // K tile via async DMA; V tile manual (needs transpose into Vt)
      async_copy64(&Ks[lrow][lcb],
                   kh + ((size_t)(b * SEQ + j0 + lrow)) * DMODEL + n * DHEAD + lcb);
      const bh8* gv = (const bh8*)(vh + ((size_t)(b * SEQ + j0 + lrow)) * DMODEL + n * DHEAD + lcb);
      bh8 v0 = gv[0], v1 = gv[1], v2 = gv[2], v3 = gv[3];
#pragma unroll
      for (int e = 0; e < 8; ++e) {
        Vt[lcb + e][lrow]      = v0[e];
        Vt[lcb + 8 + e][lrow]  = v1[e];
        Vt[lcb + 16 + e][lrow] = v2[e];
        Vt[lcb + 24 + e][lrow] = v3[e];
      }
      if (j0 + 64 < SEQ)  // prefetch next key tile -> global_prefetch_b8
        __builtin_prefetch(kh + ((size_t)(b * SEQ + j0 + 64 + lrow)) * DMODEL + n * DHEAD, 0, 1);
    }
    wait_async0();
    __syncthreads();

    // content scores S = Q @ K^T (64 keys per wave row-block)
    v8f sc[4];
#pragma unroll
    for (int f = 0; f < 4; ++f) {
      v16bf b0 = frag_b(&Ks[16 * f][0], 72, lane);
      v16bf b1 = frag_b(&Ks[16 * f][32], 72, lane);
      v8f s = {};
      s = wmma_bf16(qa[0], b0, s);
      s = wmma_bf16(qa[1], b1, s);
      sc[f] = s;
    }

    // + disentangled bias (gather from c2p/p2c) + mask
#pragma unroll
    for (int f = 0; f < 4; ++f) {
      const int j = j0 + 16 * f + (lane & 15);
      const float mterm = -1e6f * (1.0f - mask[b * SEQ + j]);
#pragma unroll
      for (int r = 0; r < 8; ++r) {
        const int i = i0 + 16 * wave + rhalf * 8 + r;
        float sv = sc[f][r] + mterm;
        int rel = i - j + 512;
        rel = rel < 0 ? 0 : (rel > 1023 ? 1023 : rel);
        if (i > 0 && j > 0)
          sv += (float)c2p[relbase + (size_t)i * SPAN2 + rel] +
                (float)p2c[relbase + (size_t)j * SPAN2 + rel];
        sc[f][r] = sv;
      }
    }

    // online softmax (rows live in one VGPR slot across a 16-lane half)
    float mloc[8];
#pragma unroll
    for (int r = 0; r < 8; ++r)
      mloc[r] = fmaxf(fmaxf(sc[0][r], sc[1][r]), fmaxf(sc[2][r], sc[3][r]));
#pragma unroll
    for (int xm = 1; xm <= 8; xm <<= 1)
#pragma unroll
      for (int r = 0; r < 8; ++r)
        mloc[r] = fmaxf(mloc[r], __shfl_xor(mloc[r], xm, 32));
    float psum[8];
#pragma unroll
    for (int r = 0; r < 8; ++r) {
      const float mnew = fmaxf(mrun[r], mloc[r]);
      const float al = __expf(mrun[r] - mnew);
      mrun[r] = mnew;
      lrun[r] *= al;
#pragma unroll
      for (int fd = 0; fd < 4; ++fd) acc_o[fd][r] *= al;
      psum[r] = 0.0f;
    }
#pragma unroll
    for (int f = 0; f < 4; ++f)
#pragma unroll
      for (int r = 0; r < 8; ++r) {
        const float p = __expf(sc[f][r] - mrun[r]);
        psum[r] += p;
        Ps[16 * wave + rhalf * 8 + r][16 * f + (lane & 15)] = (bf16_t)p;
      }
#pragma unroll
    for (int xm = 1; xm <= 8; xm <<= 1)
#pragma unroll
      for (int r = 0; r < 8; ++r) psum[r] += __shfl_xor(psum[r], xm, 32);
#pragma unroll
    for (int r = 0; r < 8; ++r) lrun[r] += psum[r];
    __syncthreads();

    // O += P @ V   (P fragments are fd-invariant: load once per key block)
    v16bf pa0 = frag_a(&Ps[16 * wave][0], 72, lane);
    v16bf pa1 = frag_a(&Ps[16 * wave][32], 72, lane);
#pragma unroll
    for (int fd = 0; fd < 4; ++fd) {
      v16bf vb0 = frag_b(&Vt[16 * fd][0], 72, lane);
      v16bf vb1 = frag_b(&Vt[16 * fd][32], 72, lane);
      acc_o[fd] = wmma_bf16(pa0, vb0, acc_o[fd]);
      acc_o[fd] = wmma_bf16(pa1, vb1, acc_o[fd]);
    }
  }

#pragma unroll
  for (int fd = 0; fd < 4; ++fd)
#pragma unroll
    for (int r = 0; r < 8; ++r) {
      const int i = i0 + 16 * wave + rhalf * 8 + r;
      attn_out[((size_t)(b * SEQ + i)) * DMODEL + n * DHEAD + 16 * fd + (lane & 15)] =
          acc_o[fd][r] / lrun[r];
    }
}

// ---------------- Kernel 4: residual + LayerNorm ----------------
__global__ __launch_bounds__(256)
void ln_residual_kernel(const float* __restrict__ attn, const float* __restrict__ query,
                        const float* __restrict__ gamma, const float* __restrict__ beta,
                        float* __restrict__ out) {
  __shared__ float red[8];
  const int t = threadIdx.x;
  const size_t base = (size_t)blockIdx.x * DMODEL;
  float v0 = attn[base + t], v1 = attn[base + t + 256], v2 = attn[base + t + 512];
  float s = v0 + v1 + v2;
#pragma unroll
  for (int xm = 1; xm < 32; xm <<= 1) s += __shfl_xor(s, xm, 32);
  if ((t & 31) == 0) red[t >> 5] = s;
  __syncthreads();
  s = 0.0f;
#pragma unroll
  for (int w = 0; w < 8; ++w) s += red[w];
  const float mu = s * (1.0f / DMODEL);
  const float d0 = v0 - mu, d1 = v1 - mu, d2 = v2 - mu;
  float q2 = d0 * d0 + d1 * d1 + d2 * d2;
#pragma unroll
  for (int xm = 1; xm < 32; xm <<= 1) q2 += __shfl_xor(q2, xm, 32);
  __syncthreads();
  if ((t & 31) == 0) red[t >> 5] = q2;
  __syncthreads();
  q2 = 0.0f;
#pragma unroll
  for (int w = 0; w < 8; ++w) q2 += red[w];
  const float rstd = rsqrtf(q2 * (1.0f / DMODEL) + 1e-5f);
  out[base + t]       = query[base + t]       + d0 * rstd * gamma[t]       + beta[t];
  out[base + t + 256] = query[base + t + 256] + d1 * rstd * gamma[t + 256] + beta[t + 256];
  out[base + t + 512] = query[base + t + 512] + d2 * rstd * gamma[t + 512] + beta[t + 512];
}

// ---------------- host launcher ----------------
extern "C" void kernel_launch(void* const* d_in, const int* in_sizes, int n_in,
                              void* d_out, int out_size, void* d_ws, size_t ws_size,
                              hipStream_t stream) {
  (void)in_sizes; (void)n_in; (void)out_size; (void)ws_size;
  const float* query  = (const float*)d_in[0];
  const float* amask  = (const float*)d_in[1];
  const float* posk_e = (const float*)d_in[2];
  const float* posq_e = (const float*)d_in[3];
  const float* Wq  = (const float*)d_in[4];
  const float* bq  = (const float*)d_in[5];
  const float* Wk  = (const float*)d_in[6];
  const float* Wv  = (const float*)d_in[7];
  const float* bv  = (const float*)d_in[8];
  const float* Wpq = (const float*)d_in[9];
  const float* bpq = (const float*)d_in[10];
  const float* Wpk = (const float*)d_in[11];
  const float* rwb = (const float*)d_in[12];
  const float* gam = (const float*)d_in[13];
  const float* bet = (const float*)d_in[14];
  float* out = (float*)d_out;

  // workspace carve-out (bf16 intermediates; c2p/p2c dominate at ~100 MB each)
  size_t off = 0;
  auto carve = [&](size_t bytes) -> void* {
    void* p = (char*)d_ws + off;
    off += (bytes + 255) & ~(size_t)255;
    return p;
  };
  const size_t MS = (size_t)BATCH * SEQ;                 // 4096
  bf16_t* qh_w   = (bf16_t*)carve(MS * DMODEL * 2);
  bf16_t* kh_w   = (bf16_t*)carve(MS * DMODEL * 2);
  bf16_t* vh_w   = (bf16_t*)carve(MS * DMODEL * 2);
  bf16_t* posk_w = (bf16_t*)carve((size_t)SPAN2 * DMODEL * 2);
  bf16_t* posq_w = (bf16_t*)carve((size_t)SPAN2 * DMODEL * 2);
  bf16_t* c2p_w  = (bf16_t*)carve((size_t)BATCH * NHEAD * SEQ * SPAN2 * 2);
  bf16_t* p2c_w  = (bf16_t*)carve((size_t)BATCH * NHEAD * SEQ * SPAN2 * 2);
  float*  attn_w = (float*)carve(MS * DMODEL * 4);

  const dim3 blk(128);
  // projections: qh = (xWq^T + bq + r_w_bias)*SCALE ; kh = xWk^T ; v = xWv^T + bv
  gemm_proj_kernel<<<dim3(MS / 64, DMODEL / 64), blk, 0, stream>>>(
      query, DMODEL, Wq, DMODEL, bq, rwb, SCALE_F, qh_w, DMODEL, DMODEL);
  gemm_proj_kernel<<<dim3(MS / 64, DMODEL / 64), blk, 0, stream>>>(
      query, DMODEL, Wk, DMODEL, nullptr, nullptr, 1.0f, kh_w, DMODEL, DMODEL);
  gemm_proj_kernel<<<dim3(MS / 64, DMODEL / 64), blk, 0, stream>>>(
      query, DMODEL, Wv, DMODEL, bv, nullptr, 1.0f, vh_w, DMODEL, DMODEL);
  // pos_k = posk_e Wpk^T ; pos_q_scaled = (posq_e Wpq^T + bpq) * SCALE  (folds p2c k-scale)
  gemm_proj_kernel<<<dim3(SPAN2 / 64, DMODEL / 64), blk, 0, stream>>>(
      posk_e, DMODEL, Wpk, DMODEL, nullptr, nullptr, 1.0f, posk_w, DMODEL, DMODEL);
  gemm_proj_kernel<<<dim3(SPAN2 / 64, DMODEL / 64), blk, 0, stream>>>(
      posq_e, DMODEL, Wpq, DMODEL, bpq, nullptr, SCALE_F, posq_w, DMODEL, DMODEL);

  // per-head relative tables: c2p = qh @ pos_k^T ; p2c = kh @ pos_q_scaled^T
  gemm_rel_kernel<<<dim3(SEQ / 64, SPAN2 / 64, BATCH * NHEAD), blk, 0, stream>>>(
      qh_w, posk_w, c2p_w);
  gemm_rel_kernel<<<dim3(SEQ / 64, SPAN2 / 64, BATCH * NHEAD), blk, 0, stream>>>(
      kh_w, posq_w, p2c_w);

  // fused attention with online softmax
  attn_kernel<<<dim3(SEQ / 64, BATCH * NHEAD), blk, 0, stream>>>(
      qh_w, kh_w, vh_w, c2p_w, p2c_w, amask, attn_w);

  // residual + LayerNorm
  ln_residual_kernel<<<dim3(MS), dim3(256), 0, stream>>>(attn_w, query, gam, bet, out);
}
